// MoE_49074296324578
// MI455X (gfx1250) — matrix operand order repaired and checked by reading, not compiled, not run on hardware
//
#include <hip/hip_runtime.h>
#include <cstdint>
#include <cstddef>

// ---------------------------------------------------------------------------
// MoE forward for MI455X (gfx1250, wave32, WMMA bf16 16x16x32, f32 accum)
// Weights are pre-swizzled into WMMA B-fragment order so the GEMM inner loop
// is b128 loads (SGPR base + 32-bit VGPR offset) + v_wmma only.
// ---------------------------------------------------------------------------

typedef __attribute__((ext_vector_type(16))) __bf16 v16bf;
typedef __attribute__((ext_vector_type(8)))  float  v8f;

static constexpr int   T_ = 8192;    // B*S
static constexpr int   D_ = 512;
static constexpr int   F_ = 1024;
static constexpr int   E_ = 8;
static constexpr int   K_ = 2;
static constexpr int   TM = 32;      // tokens per FFN tile
static constexpr int   FN = 64;      // F chunk
static constexpr float LB_W = 0.01f;
static constexpr float Z_W  = 0.001f;

// workspace layout (bytes); every section is 16B-aligned
static constexpr size_t O_CNT = 0;                       // int  cnt[E]
static constexpr size_t O_P   = 64;                      // f32  P_sum[E]
static constexpr size_t O_Z   = 128;                     // f32  z_sum
static constexpr size_t O_IDX = 256;                     // int  idx[E][T]
static constexpr size_t O_WGT = O_IDX + (size_t)E_*T_*4; // f32  wgt[E][T]
static constexpr size_t O_XB  = O_WGT + (size_t)E_*T_*4; // bf16 x[T][D]
static constexpr size_t O_GW  = O_XB + (size_t)T_*D_*2;  // bf16 gate_w swizzled
static constexpr size_t O_UW  = O_GW + (size_t)E_*D_*F_*2;
static constexpr size_t O_DW  = O_UW + (size_t)E_*D_*F_*2;

// ---------------------------------------------------------------------------
__global__ void zero_f32_kernel(float* __restrict__ p, size_t n) {
  size_t i = (size_t)blockIdx.x * blockDim.x + threadIdx.x;
  size_t stride = (size_t)gridDim.x * blockDim.x;
  for (; i < n; i += stride) p[i] = 0.0f;
}

__global__ void f32_to_bf16_kernel(const float* __restrict__ src,
                                   __bf16* __restrict__ dst, size_t n) {
  size_t i = (size_t)blockIdx.x * blockDim.x + threadIdx.x;
  size_t stride = (size_t)gridDim.x * blockDim.x;
  for (; i < n; i += stride) dst[i] = (__bf16)src[i];
}

// ---------------------------------------------------------------------------
// Pack f32 [e][KDIM][NDIM] row-major weights into bf16 WMMA B-fragment order:
//   dst[(((kbi*NT + nt)*32 + lane)*16 + j] =
//     src[(kbi*32 + (j&8)*2 + (lane>>4)*8 + (j&7)) * NDIM + nt*16 + (lane&15)]
// so each lane's 16 B-fragment elements are 32 contiguous bytes (2x b128).
// Compile-time dims -> all index math folds to shifts/masks.
// ---------------------------------------------------------------------------
template <int KDIM, int NDIM>
__global__ void pack_b_kernel(const float* __restrict__ src,
                              __bf16* __restrict__ dst) {
  constexpr int NT = NDIM >> 4;
  constexpr size_t per = (size_t)KDIM * NDIM;
  constexpr size_t total = (size_t)E_ * per;
  size_t i = (size_t)blockIdx.x * blockDim.x + threadIdx.x;
  const size_t stride = (size_t)gridDim.x * blockDim.x;
  for (; i < total; i += stride) {
    const size_t e = i / per;                 // per is a power of two
    const size_t o = i - e * per;
    const int j    = (int)(o & 15);
    const int lane = (int)((o >> 4) & 31);
    const int nt   = (int)((o >> 9) & (NT - 1));
    const int kbi  = (int)(o / ((size_t)NT * 512));
    const int k = kbi * 32 + (j & 8) * 2 + (lane >> 4) * 8 + (j & 7);
    const int n = nt * 16 + (lane & 15);
    dst[i] = (__bf16)src[e * per + (size_t)k * NDIM + n];
  }
}

// ---------------------------------------------------------------------------
// Router: fp32, one wave (32 lanes) per token. Builds per-expert token lists
// and accumulates aux-loss statistics.
// ---------------------------------------------------------------------------
__global__ __launch_bounds__(256)
void router_kernel(const float* __restrict__ x, const float* __restrict__ gate_W,
                   int* __restrict__ cnt, int* __restrict__ idx,
                   float* __restrict__ wgt, float* __restrict__ P_sum,
                   float* __restrict__ z_sum) {
  const int wave = threadIdx.x >> 5;
  const int lane = threadIdx.x & 31;
  const int t = blockIdx.x * 8 + wave;
  if (t >= T_) return;

  const float* xr = x + (size_t)t * D_;
  float logits[E_];
#pragma unroll
  for (int e = 0; e < E_; ++e) {
    const float* w = gate_W + (size_t)e * D_;
    float s = 0.0f;
    for (int d = lane; d < D_; d += 32) s += xr[d] * w[d];
#pragma unroll
    for (int off = 16; off > 0; off >>= 1) s += __shfl_xor(s, off, 32);
    logits[e] = s;                       // all lanes hold the full logit vector
  }

  float m = logits[0];
#pragma unroll
  for (int e = 1; e < E_; ++e) m = fmaxf(m, logits[e]);
  float p[E_];
  float se = 0.0f;
#pragma unroll
  for (int e = 0; e < E_; ++e) { p[e] = __expf(logits[e] - m); se += p[e]; }
  const float inv = 1.0f / se;
#pragma unroll
  for (int e = 0; e < E_; ++e) p[e] *= inv;
  const float lse = m + __logf(se);

  // top-2 (first occurrence wins ties, like lax.top_k)
  int i1 = 0; float v1 = p[0];
#pragma unroll
  for (int e = 1; e < E_; ++e) if (p[e] > v1) { v1 = p[e]; i1 = e; }
  int i2 = -1; float v2 = -1.0f;
#pragma unroll
  for (int e = 0; e < E_; ++e) {
    if (e == i1) continue;
    if (p[e] > v2) { v2 = p[e]; i2 = e; }
  }
  const float s2 = v1 + v2;
  const float w1 = v1 / s2, w2 = v2 / s2;

  if (lane == 0) {
    int pos1 = atomicAdd(&cnt[i1], 1);
    idx[(size_t)i1 * T_ + pos1] = t;
    wgt[(size_t)i1 * T_ + pos1] = w1;
    int pos2 = atomicAdd(&cnt[i2], 1);
    idx[(size_t)i2 * T_ + pos2] = t;
    wgt[(size_t)i2 * T_ + pos2] = w2;
    atomicAdd(z_sum, lse * lse);
  }
  if (lane < E_) atomicAdd(&P_sum[lane], p[lane]);
}

// ---------------------------------------------------------------------------
// Fragment loads. Global variants take (uniform SGPR base, 32-bit byte offset)
// so the backend can use GVS addressing: global_load_b128 v, voff, s[base].
// ---------------------------------------------------------------------------
__device__ __forceinline__ v16bf load_frag_lds(const __bf16* rp) {
  v16bf a;
  ((uint4*)&a)[0] = *(const uint4*)rp;
  ((uint4*)&a)[1] = *(const uint4*)(rp + 16);
  return a;
}

__device__ __forceinline__ v16bf load_frag_g(const __bf16* base, int byteOff) {
  v16bf b;
  ((uint4*)&b)[0] = *(const uint4*)((const char*)base + byteOff);
  ((uint4*)&b)[1] = *(const uint4*)((const char*)base + byteOff + 16);
  return b;
}

// ---------------------------------------------------------------------------
// Per-expert fused FFN: out[tok] += w * ( silu(X Gw) * (X Uw) ) Dw
// grid = (T/TM tiles, E experts), block = 256 threads = 8 waves.
// ---------------------------------------------------------------------------
__global__ __launch_bounds__(256)
void moe_ffn_kernel(const __bf16* __restrict__ xb, const __bf16* __restrict__ gw,
                    const __bf16* __restrict__ uw, const __bf16* __restrict__ dw,
                    const int* __restrict__ cnt, const int* __restrict__ idx,
                    const float* __restrict__ wgt, float* __restrict__ out) {
  const int e = blockIdx.y;
  const int n = cnt[e];
  const int row0 = blockIdx.x * TM;
  if (row0 >= n) return;                   // uniform per block

  __shared__ __bf16 Xs[TM][D_ + 8];        // +8 bf16 pad: rows shift 4 banks
  __shared__ __bf16 Hs[TM][FN + 8];
  __shared__ int   toks[TM];
  __shared__ float twgt[TM];

  const int tid  = threadIdx.x;
  const int lane = tid & 31;
  const int wv   = tid >> 5;
  const int lhalf = lane >> 4;             // 0 | 1
  const int lmod  = lane & 15;

  if (tid < TM) {
    int r = row0 + tid;
    int tok = 0; float w = 0.0f;
    if (r < n) { tok = idx[(size_t)e * T_ + r]; w = wgt[(size_t)e * T_ + r]; }
    toks[tid] = tok; twgt[tid] = w;
  }
  __syncthreads();

  // gather token rows into LDS (16B vector copies; base + i32 byte offset)
  for (int i = tid; i < TM * (D_ / 8); i += 256) {
    int r = i / (D_ / 8);
    int c = (i % (D_ / 8)) * 8;
    int srcOff = toks[r] * (D_ * 2) + c * 2;          // bytes, < 8 MB
    *(uint4*)&Xs[r][c] = *(const uint4*)((const char*)xb + srcOff);
  }
  __syncthreads();

  // phase-A tile ownership: 2 (M) x 4 (N) tiles over [TM x FN]
  const int m_a = (wv >> 2) * 16;
  const int nt_a_off = (wv & 3);           // tile index offset inside FN chunk
  // phase-B tile ownership: 2 (M) x 32 (N) tiles over [TM x D], 8 tiles/wave
  const int m_b = (wv & 1) * 16;
  const int nb0 = (wv >> 1) * 8;

  const int NT_G = F_ >> 4;                // 64 N-tiles for gate/up
  const int NT_D = D_ >> 4;                // 32 N-tiles for down

  // uniform per-expert base pointers (stay in SGPRs)
  const __bf16* gwe = gw + (size_t)e * D_ * F_;
  const __bf16* uwe = uw + (size_t)e * D_ * F_;
  const __bf16* dwe = dw + (size_t)e * F_ * D_;

  v8f oacc[8];
#pragma unroll
  for (int j = 0; j < 8; ++j) oacc[j] = (v8f){};

  for (int fc = 0; fc < F_; fc += FN) {
    // ---- phase A: Hg = X Gw[:,fc:fc+FN], Hu = X Uw[:,fc:fc+FN]
    const int nt_a = (fc >> 4) + nt_a_off; // this wave's N-tile (gate/up)
    v8f accg = (v8f){}, accu = (v8f){};
#pragma unroll 4
    for (int kb = 0; kb < D_; kb += 32) {
      const int kbi = kb >> 5;
      const v16bf a = load_frag_lds(&Xs[m_a + lmod][kb + lhalf * 8]);
      // fragment byte offset, fits i32 (per-expert region is 1 MB)
      const int fo = (((kbi * NT_G + nt_a) << 5) + lane) << 5;
      const v16bf bg = load_frag_g(gwe, fo);
      const v16bf bu = load_frag_g(uwe, fo);
      accg = __builtin_amdgcn_wmma_f32_16x16x32_bf16(false, a, false, bg,
                                                     (short)0, accg, false, false);
      accu = __builtin_amdgcn_wmma_f32_16x16x32_bf16(false, a, false, bu,
                                                     (short)0, accu, false, false);
    }

    __syncthreads();       // previous chunk's phase-B reads of Hs are done
#pragma unroll
    for (int r = 0; r < 8; ++r) {
      float g = accg[r], u = accu[r];
      // silu(g)*u with fast v_rcp_f32 (no IEEE div refinement chain)
      float h = g * __builtin_amdgcn_rcpf(1.0f + __expf(-g)) * u;
      Hs[m_a + r + lhalf * 8][(nt_a_off << 4) + lmod] = (__bf16)h;
    }
    __syncthreads();

    // ---- phase B: out += H @ Dw[fc:fc+FN, :]
#pragma unroll
    for (int ks = 0; ks < FN; ks += 32) {
      const int kbi = (fc + ks) >> 5;
      const v16bf ah = load_frag_lds(&Hs[m_b + lmod][ks + lhalf * 8]);
#pragma unroll
      for (int j = 0; j < 8; ++j) {
        const int fo = (((kbi * NT_D + nb0 + j) << 5) + lane) << 5;
        const v16bf bd = load_frag_g(dwe, fo);
        oacc[j] = __builtin_amdgcn_wmma_f32_16x16x32_bf16(false, ah, false, bd,
                                                          (short)0, oacc[j], false, false);
      }
    }
  }

  // ---- scatter with routing weight (K=2 lists may collide -> atomic f32 add)
#pragma unroll
  for (int j = 0; j < 8; ++j) {
#pragma unroll
    for (int r = 0; r < 8; ++r) {
      const int rr = m_b + r + lhalf * 8;
      const float w = twgt[rr];
      if (w != 0.0f) {
        const int outOff = toks[rr] * (D_ * 4) + ((nb0 + j) * 16 + lmod) * 4;
        atomicAdd((float*)((char*)out + outOff), w * oacc[j][r]);
      }
    }
  }
}

// ---------------------------------------------------------------------------
__global__ void finalize_kernel(const int* __restrict__ cnt,
                                const float* __restrict__ P_sum,
                                const float* __restrict__ z_sum,
                                float* __restrict__ aux_out) {
  if (threadIdx.x == 0 && blockIdx.x == 0) {
    float lb = 0.0f;
#pragma unroll
    for (int e = 0; e < E_; ++e) {
      float f = (float)cnt[e] / (float)(T_ * K_);
      float P = P_sum[e] / (float)T_;
      lb += f * P;
    }
    lb *= (float)E_;
    float z = z_sum[0] / (float)T_;
    aux_out[0] = LB_W * lb + Z_W * z;
  }
}

// ---------------------------------------------------------------------------
extern "C" void kernel_launch(void* const* d_in, const int* in_sizes, int n_in,
                              void* d_out, int out_size, void* d_ws, size_t ws_size,
                              hipStream_t stream) {
  (void)in_sizes; (void)n_in; (void)out_size; (void)ws_size;

  const float* x      = (const float*)d_in[0];   // [B,S,D]
  const float* gate_W = (const float*)d_in[1];   // [E,D]
  const float* gate_w = (const float*)d_in[2];   // [E,D,F]
  const float* up_w   = (const float*)d_in[3];   // [E,D,F]
  const float* down_w = (const float*)d_in[4];   // [E,F,D]
  float* out = (float*)d_out;                    // [B,S,D] ++ [aux]

  char* ws = (char*)d_ws;
  int*    cnt   = (int*)(ws + O_CNT);
  float*  P_sum = (float*)(ws + O_P);
  float*  z_sum = (float*)(ws + O_Z);
  int*    idx   = (int*)(ws + O_IDX);
  float*  wgt   = (float*)(ws + O_WGT);
  __bf16* xb    = (__bf16*)(ws + O_XB);
  __bf16* gwb   = (__bf16*)(ws + O_GW);
  __bf16* uwb   = (__bf16*)(ws + O_UW);
  __bf16* dwb   = (__bf16*)(ws + O_DW);

  const size_t nOut = (size_t)T_ * D_;

  // 1) zero output accumulator + ws header (counts / P / z)
  zero_f32_kernel<<<2048, 256, 0, stream>>>(out, nOut);
  zero_f32_kernel<<<1, 64, 0, stream>>>((float*)ws, 64);

  // 2) bf16 conversion: x flat; weights swizzled to WMMA B-fragment order
  f32_to_bf16_kernel<<<2048, 256, 0, stream>>>(x, xb, (size_t)T_ * D_);
  pack_b_kernel<D_, F_><<<4096, 256, 0, stream>>>(gate_w, gwb);
  pack_b_kernel<D_, F_><<<4096, 256, 0, stream>>>(up_w,   uwb);
  pack_b_kernel<F_, D_><<<4096, 256, 0, stream>>>(down_w, dwb);

  // 3) fp32 router + top-2 + per-expert token lists + aux stats
  router_kernel<<<T_ / 8, 256, 0, stream>>>(x, gate_W, cnt, idx, wgt, P_sum, z_sum);

  // 4) routed expert FFN (worst-case tile count per expert; excess blocks exit)
  dim3 grid(T_ / TM, E_);
  moe_ffn_kernel<<<grid, 256, 0, stream>>>(xb, gwb, uwb, dwb, cnt, idx, wgt, out);

  // 5) aux loss scalar
  finalize_kernel<<<1, 32, 0, stream>>>(cnt, P_sum, z_sum, out + nOut);
}
